// GetEdgeFeature_69080253989617
// MI455X (gfx1250) — matrix operand order repaired and checked by reading, not compiled, not run on hardware
//
#include <hip/hip_runtime.h>
#include <hip/hip_bf16.h>

#define BB 8
#define NN 2048
#define CC 64
#define IN_CH 128
#define OUT_CH 256
#define KNN 16
#define NPTS (BB * NN)          // 16384
#define EPSC 1e-5f
#define SLOPE 0.2f
#define FLTMAX 3.402823466e38f

typedef __attribute__((ext_vector_type(16))) __bf16 v16bf;
typedef __attribute__((ext_vector_type(8)))  float  v8f;

// ---------------------------------------------------------------- WMMA helpers

__device__ __forceinline__ v8f wmma_bf16(v16bf a, v16bf b, v8f c) {
  // emits v_wmma_f32_16x16x32_bf16
  return __builtin_amdgcn_wmma_f32_16x16x32_bf16(
      /*neg_a=*/false, a, /*neg_b=*/false, b,
      /*c_mod=*/(short)0, c, /*reuse_a=*/false, /*reuse_b=*/false);
}

__device__ __forceinline__ v8f v8f_zero() {
  v8f c;
#pragma unroll
  for (int i = 0; i < 8; ++i) c[i] = 0.0f;
  return c;
}

// A fragment (16-bit A 16x32): element e -> K = kbase + 16*half + e
__device__ __forceinline__ v16bf afrag_from_bf16(const __bf16* row, int kbase, int half) {
  v16bf f;
#pragma unroll
  for (int e = 0; e < 16; ++e) {
    f[e] = row[kbase + 16 * half + e];
  }
  return f;
}
// B fragment (16-bit B 32x16): element e -> K = kbase + 16*half + e  -> one aligned 32B load
__device__ __forceinline__ v16bf bfrag_load(const __bf16* row, int kbase, int half) {
  return *(const v16bf*)(row + kbase + 16 * half);
}

// LDS byte offset of a generic (shared) pointer: low 32 bits of the flat address
__device__ __forceinline__ unsigned lds_off(const void* p) {
  return (unsigned)(unsigned long long)p;
}

// ---------------------------------------------------------------- kernel 0: init
// zero BN accumulators, bf16-convert weights AND points, per-point squared norms
__global__ void init_kernel(const float* __restrict__ pts,
                            const float* __restrict__ l1w, const float* __restrict__ convw,
                            __bf16* __restrict__ pts_bf,
                            __bf16* __restrict__ l1w_bf, __bf16* __restrict__ convw_bf,
                            float* __restrict__ sqn, float* __restrict__ stats) {
  const int i = blockIdx.x * blockDim.x + threadIdx.x;   // grid covers NPTS*CC
  if (i < 2 * OUT_CH) stats[i] = 0.0f;
  if (i < IN_CH * IN_CH) l1w_bf[i] = (__bf16)l1w[i];
  if (i < OUT_CH * IN_CH) convw_bf[i] = (__bf16)convw[i];
  if (i < NPTS) {
    const float* p = pts + (size_t)i * CC;
    float s = 0.0f;
#pragma unroll
    for (int c = 0; c < CC; ++c) s += p[c] * p[c];
    sqn[i] = s;
  }
  if (i < NPTS * CC) pts_bf[i] = (__bf16)pts[i];
}

// ---------------------------------------------------------------- kernel 1: kNN
// block = 16 query rows, 16 waves. bf16 WMMA inner products -> 16x2048 LDS
// distance stripe -> one wave per row does 32 ordered min-extractions.
__global__ void __launch_bounds__(512)
knn_kernel(const __bf16* __restrict__ pts_bf, const float* __restrict__ sqn,
           int* __restrict__ nn_idx) {
  const int b    = blockIdx.x >> 7;          // N/16 = 128 row-tiles per batch
  const int row0 = (blockIdx.x & 127) << 4;
  extern __shared__ float dist[];            // 16*2048 f32 = 128 KB

  const int tid  = threadIdx.x;
  const int wave = tid >> 5, lane = tid & 31;
  const int half = lane >> 4, l15 = lane & 15;

  // A fragments straight from the bf16 point buffer (row M = l15)
  const __bf16* qrow = pts_bf + (size_t)(b * NN + row0 + l15) * CC;
  v16bf aQ[2];
#pragma unroll
  for (int s = 0; s < 2; ++s) aQ[s] = afrag_from_bf16(qrow, 32 * s, half);
  float sqq[8];
#pragma unroll
  for (int r = 0; r < 8; ++r) sqq[r] = sqn[b * NN + row0 + r + 8 * half];

  for (int t = wave; t < NN / 16; t += 16) {
    const int col = t * 16 + l15;
    const __bf16* prow = pts_bf + (size_t)(b * NN + col) * CC;
    if (t + 16 < NN / 16)  // hide L2 latency of the next B tile
      __builtin_prefetch(pts_bf + (size_t)(b * NN + (t + 16) * 16 + l15) * CC, 0, 3);
    v8f c = v8f_zero();
#pragma unroll
    for (int s = 0; s < 2; ++s)
      c = wmma_bf16(aQ[s], bfrag_load(prow, 32 * s, half), c);
    const float sqp = sqn[b * NN + col];
#pragma unroll
    for (int r = 0; r < 8; ++r)
      dist[(r + 8 * half) * NN + col] = sqq[r] - 2.0f * c[r] + sqp;
  }
  __syncthreads();

  // wave `wave` selects for query row M = wave
  float* drow = dist + wave * NN;
  int* outp = nn_idx + (size_t)(b * NN + row0 + wave) * KNN;
  for (int it = 0; it < 32; ++it) {
    float best = FLTMAX;
    int bi = 0x7fffffff;
    for (int j = lane; j < NN; j += 32) {
      float v = drow[j];
      if (v < best) { best = v; bi = j; }       // strict: keeps lowest index
    }
#pragma unroll
    for (int off = 16; off; off >>= 1) {
      float ob = __shfl_xor(best, off);
      int   oi = __shfl_xor(bi, off);
      if (ob < best || (ob == best && oi < bi)) { best = ob; bi = oi; }
    }
    if ((bi & 31) == lane) drow[bi] = FLTMAX;   // remove extracted candidate
    if (lane == 0 && !(it & 1)) outp[it >> 1] = bi;   // nn_idx[..., ::2]
  }
}

// ---------------------------------------------------------------- kernel 2: attention
// block = one point; neighbors DMA'd into LDS with async-to-LDS; 8 waves = 8
// output-channel tiles of att = feat @ l1_w^T; softmax over M in-register.
__global__ void __launch_bounds__(256)
attn_kernel(const float* __restrict__ pts, const int* __restrict__ nn_idx,
            const __bf16* __restrict__ l1w_bf, const float* __restrict__ l1b,
            float* __restrict__ fagg) {
  const int p = blockIdx.x;
  const int b = p >> 11;                       // N = 2048
  __shared__ float  edge[KNN * CC];            // async-staged neighbor rows (f32)
  __shared__ float  feat[16 * IN_CH];
  __shared__ __bf16 featb[16 * IN_CH];
  __shared__ int    idx[KNN];
  const int tid = threadIdx.x;
  if (tid < KNN) idx[tid] = nn_idx[(size_t)p * KNN + tid];
  __syncthreads();

  // DMA the 16 scattered 256B neighbor rows into LDS: 256 lanes x 16B each.
  {
    const int row = tid >> 4, ch = tid & 15;
    const unsigned ldsa = lds_off(edge) + (unsigned)(row * CC + ch * 4) * 4u;
    const unsigned long long ga =
        (unsigned long long)(pts + ((size_t)b * NN + idx[row]) * CC + ch * 4);
    asm volatile("global_load_async_to_lds_b128 %0, %1, off"
                 :: "v"(ldsa), "v"(ga) : "memory");
  }
  asm volatile("s_wait_asynccnt 0" ::: "memory");
  __syncthreads();

  const float* center = pts + (size_t)p * CC;
  for (int i = tid; i < 16 * IN_CH; i += 256) {
    const int kk = i >> 7, c = i & 127;
    float v = (c < CC) ? center[c]
                       : center[c - CC] - edge[kk * CC + (c - CC)];
    feat[i]  = v;
    featb[i] = (__bf16)v;
  }
  __syncthreads();

  const int wave = tid >> 5, lane = tid & 31;
  const int half = lane >> 4, l15 = lane & 15;
  const int d0 = wave << 4;

  v8f c = v8f_zero();
#pragma unroll
  for (int s = 0; s < 4; ++s) {
    v16bf a = afrag_from_bf16(featb + l15 * IN_CH, 32 * s, half);
    v16bf w = bfrag_load(l1w_bf + (size_t)(d0 + l15) * IN_CH, 32 * s, half);
    c = wmma_bf16(a, w, c);
  }
  // + bias, softmax over the 16 neighbors (M dim: 8 local rows + lane^16)
  const float bias = l1b[d0 + l15];
  float mx = -FLTMAX;
#pragma unroll
  for (int r = 0; r < 8; ++r) { c[r] += bias; mx = fmaxf(mx, c[r]); }
  mx = fmaxf(mx, __shfl_xor(mx, 16));
  float sum = 0.0f;
#pragma unroll
  for (int r = 0; r < 8; ++r) { c[r] = __expf(c[r] - mx); sum += c[r]; }
  sum += __shfl_xor(sum, 16);
  const float inv = 1.0f / sum;

  float acc = 0.0f;
#pragma unroll
  for (int r = 0; r < 8; ++r)
    acc += feat[(r + 8 * half) * IN_CH + d0 + l15] * c[r] * inv;
  acc += __shfl_xor(acc, 16);
  if (half == 0) fagg[(size_t)p * IN_CH + d0 + l15] = acc;
}

// ---------------------------------------------------------------- kernel 3: 1x1 conv + BN stats
__global__ void __launch_bounds__(256)
conv_kernel(const float* __restrict__ fagg, const __bf16* __restrict__ convw_bf,
            const float* __restrict__ convb, float* __restrict__ y,
            float* __restrict__ stats) {
  const int row0 = blockIdx.x << 4;            // 16 points per block
  __shared__ __bf16 fb[16 * IN_CH];
  const int tid = threadIdx.x;
  for (int i = tid; i < 16 * IN_CH; i += 256)
    fb[i] = (__bf16)fagg[(size_t)row0 * IN_CH + i];
  __syncthreads();

  const int wave = tid >> 5, lane = tid & 31;
  const int half = lane >> 4, l15 = lane & 15;

  v16bf a[4];
#pragma unroll
  for (int s = 0; s < 4; ++s) a[s] = afrag_from_bf16(fb + l15 * IN_CH, 32 * s, half);

  for (int t = wave; t < OUT_CH / 16; t += 8) {
    const int o0 = t << 4;
    v8f c = v8f_zero();
#pragma unroll
    for (int s = 0; s < 4; ++s)
      c = wmma_bf16(a[s], bfrag_load(convw_bf + (size_t)(o0 + l15) * IN_CH, 32 * s, half), c);
    const float bias = convb[o0 + l15];
    float s1 = 0.0f, s2 = 0.0f;
#pragma unroll
    for (int r = 0; r < 8; ++r) {
      const float v = c[r] + bias;
      y[(size_t)(row0 + r + 8 * half) * OUT_CH + o0 + l15] = v;
      s1 += v;
      s2 += v * v;
    }
    s1 += __shfl_xor(s1, 16);
    s2 += __shfl_xor(s2, 16);
    if (half == 0) {
      atomicAdd(&stats[o0 + l15], s1);
      atomicAdd(&stats[OUT_CH + o0 + l15], s2);
    }
  }
}

// ---------------------------------------------------------------- kernel 4: BN finalize + leaky ReLU
__global__ void bn_kernel(const float* __restrict__ y, const float* __restrict__ stats,
                          const float* __restrict__ gamma, const float* __restrict__ beta,
                          float* __restrict__ out) {
  const size_t i = (size_t)blockIdx.x * blockDim.x + threadIdx.x;
  const int ch = (int)(i & (OUT_CH - 1));
  const float invn = 1.0f / (float)NPTS;
  const float mean = stats[ch] * invn;
  const float var  = stats[OUT_CH + ch] * invn - mean * mean;
  const float v = gamma[ch] * (y[i] - mean) * rsqrtf(var + EPSC) + beta[ch];
  out[i] = (v >= 0.0f) ? v : SLOPE * v;
}

// ---------------------------------------------------------------- launch
extern "C" void kernel_launch(void* const* d_in, const int* in_sizes, int n_in,
                              void* d_out, int out_size, void* d_ws, size_t ws_size,
                              hipStream_t stream) {
  (void)in_sizes; (void)n_in; (void)out_size; (void)ws_size;
  const float* pts   = (const float*)d_in[0];
  const float* l1w   = (const float*)d_in[1];
  const float* l1b   = (const float*)d_in[2];
  const float* convw = (const float*)d_in[3];
  const float* convb = (const float*)d_in[4];
  const float* gamma = (const float*)d_in[5];
  const float* beta  = (const float*)d_in[6];
  // d_in[7] = k (16), d_in[8] = d (2): compile-time constants here.

  char* ws = (char*)d_ws;                       // ~29.5 MB used
  int*    nn_idx   = (int*)(ws + 0);            //  2,097,152 B
  __bf16* pts_bf   = (__bf16*)(ws + 2097152);   //  2,097,152 B
  __bf16* l1w_bf   = (__bf16*)(ws + 4194304);   //     32,768 B
  __bf16* convw_bf = (__bf16*)(ws + 4227072);   //     65,536 B
  float*  sqn      = (float*)(ws + 4292608);    //     65,536 B
  float*  stats    = (float*)(ws + 4358144);    //      2,048 B
  float*  fagg     = (float*)(ws + 4360192);    //  8,388,608 B
  float*  y        = (float*)(ws + 12748800);   // 16,777,216 B
  float*  outp     = (float*)d_out;

  init_kernel<<<(NPTS * CC) / 256, 256, 0, stream>>>(pts, l1w, convw, pts_bf,
                                                     l1w_bf, convw_bf, sqn, stats);

  const size_t knn_lds = (size_t)(16 * NN) * sizeof(float);   // 128 KB (< 320 KB WGP LDS)
  knn_kernel<<<BB * (NN / 16), 512, knn_lds, stream>>>(pts_bf, sqn, nn_idx);

  attn_kernel<<<BB * NN, 256, 0, stream>>>(pts, nn_idx, l1w_bf, l1b, fagg);

  conv_kernel<<<NPTS / 16, 256, 0, stream>>>(fagg, convw_bf, convb, y, stats);

  bn_kernel<<<(NPTS * OUT_CH) / 256, 256, 0, stream>>>(y, stats, gamma, beta, outp);
}
